// GlobalPointer_29609504539178
// MI455X (gfx1250) — matrix operand order repaired and checked by reading, not compiled, not run on hardware
//
#include <hip/hip_runtime.h>
#include <hip/hip_bf16.h>

#define HEAD_NUM 12
#define HEAD_SIZE 64
#define INPUT_DIM 768
#define BATCH 2
#define SEQ 2048
#define NPROJ (HEAD_NUM * HEAD_SIZE * 2) /* 1536 */
#define NEGV 1000000000000.0f

typedef __attribute__((ext_vector_type(16))) __bf16 v16bf;
typedef __attribute__((ext_vector_type(8)))  __bf16 v8bf;
typedef __attribute__((ext_vector_type(8)))  float  v8f;

// A-fragment (16x32 bf16): lane half 0 holds K = k0..k0+15,
// half 1 holds K = k0+16..k0+31 (caller passes base already
// offset by half*16). Two 16B loads.
static __device__ __forceinline__ v16bf load_a_frag(const __bf16* base) {
    v8bf lo = *(const v8bf*)(base);
    v8bf hi = *(const v8bf*)(base + 8);
    v16bf a;
#pragma unroll
    for (int i = 0; i < 8; ++i) { a[i] = lo[i]; a[i + 8] = hi[i]; }
    return a;
}

// ---------------------------------------------------------------------------
// Kernel 1: fp32 -> bf16 convert of x, and W -> W^T (bf16) so that GEMM B
// fragments become contiguous per-lane loads.
// ---------------------------------------------------------------------------
__global__ void convert_kernel(const float* __restrict__ x,
                               const float* __restrict__ W,
                               __bf16* __restrict__ xbf,
                               __bf16* __restrict__ wt) {
    const int NX = BATCH * SEQ * INPUT_DIM;   // 3,145,728
    const int NW = INPUT_DIM * NPROJ;         // 1,179,648
    int i = blockIdx.x * blockDim.x + threadIdx.x;
    if (i < NX) {
        xbf[i] = (__bf16)x[i];
    } else if (i < NX + NW) {
        int o = i - NX;
        int n = o / INPUT_DIM;
        int k = o % INPUT_DIM;
        wt[o] = (__bf16)W[(size_t)k * NPROJ + n];   // wt[n][k] = W[k][n]
    }
}

// ---------------------------------------------------------------------------
// Kernel 2: proj = x@W + b via bf16 WMMA, then RoPE, write q/k bf16 in
// [b*HEAD_NUM + h][seq][64] layout. One 16x16 output tile per wave.
// ---------------------------------------------------------------------------
__global__ void proj_rope_kernel(const __bf16* __restrict__ xbf,
                                 const __bf16* __restrict__ wt,
                                 const float* __restrict__ bias,
                                 __bf16* __restrict__ qbf,
                                 __bf16* __restrict__ kbf) {
    const int lane = threadIdx.x & 31;
    const int wid  = blockIdx.x * (blockDim.x >> 5) + (threadIdx.x >> 5);
    const int mtile = wid / (NPROJ / 16);
    const int ntile = wid % (NPROJ / 16);
    const int m0 = mtile * 16;
    const int n0 = ntile * 16;
    const int half = lane >> 4;
    const int l16  = lane & 15;
    const int n = n0 + l16;

    const __bf16* arow = xbf + (size_t)(m0 + l16) * INPUT_DIM;
    const __bf16* brow = wt  + (size_t)n * INPUT_DIM;

    v8f acc = {};
#pragma unroll 4
    for (int kb = 0; kb < INPUT_DIM / 32; ++kb) {
        v16bf a  = load_a_frag(arow + kb * 32 + half * 16);
        v16bf bm = *(const v16bf*)(brow + kb * 32 + half * 16);
        acc = __builtin_amdgcn_wmma_f32_16x16x32_bf16(
            false, a, false, bm, (short)0, acc, false, false);
    }

    // Epilogue: bias + RoPE. C layout: VGPR v -> row m0 + v + 8*half,
    // lane -> column n. RoPE pairs (2j, 2j+1) are adjacent lanes.
    const float bv = bias[n];
    const int dd = n & 127;        // 0..63 = q, 64..127 = k
    const int h  = n >> 7;
    const int d  = dd & 63;
    const int j  = d >> 1;
    // freq = 10000^(-2j/64) = exp(-ln(10000) * 2j/64)
    const float freq = __expf(-9.210340371976184f * (float)(2 * j) * (1.0f / 64.0f));
    const float sgn  = (d & 1) ? 1.0f : -1.0f;
    __bf16* dst = (dd < HEAD_SIZE) ? qbf : kbf;

#pragma unroll
    for (int v = 0; v < 8; ++v) {
        const int m    = m0 + v + half * 8;
        const int bidx = m / SEQ;
        const int pos  = m % SEQ;
        float val = acc[v] + bv;
        float par = __shfl_xor(val, 1, 32);           // rotate-half partner
        float s, c;
        __sincosf((float)pos * freq, &s, &c);
        float outv = val * c + sgn * par * s;         // even: v*c - p*s ; odd: v*c + p*s
        dst[((size_t)(bidx * HEAD_NUM + h) * SEQ + pos) * HEAD_SIZE + d] = (__bf16)outv;
    }
}

// ---------------------------------------------------------------------------
// Kernel 3: logits[b,h,m,n] = (Q.K^T - masks)/8 via bf16 WMMA.
// One wave owns a 16-row block of one (b,h) and sweeps all 128 n-tiles,
// reusing the A fragments (K-data streams from L2: q/k total ~12.6 MB).
// ---------------------------------------------------------------------------
__global__ void qk_logits_kernel(const __bf16* __restrict__ qbf,
                                 const __bf16* __restrict__ kbf,
                                 const int* __restrict__ mask,
                                 float* __restrict__ out) {
    const int lane  = threadIdx.x & 31;
    const int wid   = blockIdx.x * (blockDim.x >> 5) + (threadIdx.x >> 5);
    const int mtile = wid & (SEQ / 16 - 1);      // 0..127
    const int bh    = wid >> 7;                  // 0..23
    const int b     = bh / HEAD_NUM;
    const int m0    = mtile * 16;
    const int half  = lane >> 4;
    const int l16   = lane & 15;

    const __bf16* qbase = qbf + ((size_t)bh * SEQ + (m0 + l16)) * HEAD_SIZE;
    v16bf a0 = load_a_frag(qbase + 0 * 32 + half * 16);   // K = d 0..31
    v16bf a1 = load_a_frag(qbase + 1 * 32 + half * 16);   // K = d 32..63

    int maskm[8];
#pragma unroll
    for (int v = 0; v < 8; ++v) maskm[v] = mask[b * SEQ + m0 + v + half * 8];

    const __bf16* kbh   = kbf + (size_t)bh * SEQ * HEAD_SIZE;
    float*        obase = out + (size_t)bh * SEQ * SEQ;

    for (int nt = 0; nt < SEQ / 16; ++nt) {
        const int n = nt * 16 + l16;
        const __bf16* krow = kbh + (size_t)n * HEAD_SIZE;
        // B fragment (d x 16): lane half 0 -> d k0..k0+15, half 1 -> k0+16..k0+31
        v16bf b0 = *(const v16bf*)(krow + half * 16);        // d 0..31
        v16bf b1 = *(const v16bf*)(krow + 32 + half * 16);   // d 32..63

        v8f acc = {};
        acc = __builtin_amdgcn_wmma_f32_16x16x32_bf16(
            false, a0, false, b0, (short)0, acc, false, false);
        acc = __builtin_amdgcn_wmma_f32_16x16x32_bf16(
            false, a1, false, b1, (short)0, acc, false, false);

        const int mn = mask[b * SEQ + n];
#pragma unroll
        for (int v = 0; v < 8; ++v) {
            const int m = m0 + v + half * 8;
            float val = acc[v];
            if (!maskm[v]) val -= NEGV;
            if (!mn)       val -= NEGV;
            if (n < m)     val -= NEGV;     // causal: triu keeps n >= m
            obase[(size_t)m * SEQ + n] = val * 0.125f;
        }
    }
}

// ---------------------------------------------------------------------------
// Launch
// ---------------------------------------------------------------------------
extern "C" void kernel_launch(void* const* d_in, const int* in_sizes, int n_in,
                              void* d_out, int out_size, void* d_ws, size_t ws_size,
                              hipStream_t stream) {
    const float* x    = (const float*)d_in[0];   // (2, 2048, 768)
    const float* W    = (const float*)d_in[1];   // (768, 1536)
    const float* bias = (const float*)d_in[2];   // (1536,)
    const int*   mask = (const int*)d_in[3];     // (2, 2048)
    float* out = (float*)d_out;                  // (2, 12, 2048, 2048)

    char* ws = (char*)d_ws;
    __bf16* xbf = (__bf16*)(ws);                       // 4096*768   bf16 = 6,291,456 B
    __bf16* wt  = (__bf16*)(ws + 6291456);             // 1536*768   bf16 = 2,359,296 B
    __bf16* qbf = (__bf16*)(ws + 8650752);             // 24*2048*64 bf16 = 6,291,456 B
    __bf16* kbf = (__bf16*)(ws + 14942208);            // 24*2048*64 bf16 = 6,291,456 B

    {
        int total = BATCH * SEQ * INPUT_DIM + INPUT_DIM * NPROJ;  // 4,325,376
        convert_kernel<<<(total + 255) / 256, 256, 0, stream>>>(x, W, xbf, wt);
    }
    {
        int waves = (BATCH * SEQ / 16) * (NPROJ / 16);            // 256*96 = 24576
        proj_rope_kernel<<<waves / 8, 256, 0, stream>>>(xbf, wt, bias, qbf, kbf);
    }
    {
        int waves = BATCH * HEAD_NUM * (SEQ / 16);                // 24*128 = 3072
        qk_logits_kernel<<<waves / 8, 256, 0, stream>>>(qbf, kbf, mask, out);
    }
}
